// KeypointEncoder_89103391522929
// MI455X (gfx1250) — matrix operand order repaired and checked by reading, not compiled, not run on hardware
//
#include <hip/hip_runtime.h>
#include <hip/hip_bf16.h>

typedef __attribute__((ext_vector_type(16))) _Float16 v16h;
typedef __attribute__((ext_vector_type(8)))  _Float16 v8h;
typedef __attribute__((ext_vector_type(8)))  float    v8f;

#define NJ   75
#define MP   80     // M padded to 5 tiles of 16
#define KP   96     // K (joints) padded to 3 k-steps of 32
#define MID  128
#define HID  256
#define NTHREADS 256

union HV { v16h v16; v8h v8[2]; };

__device__ __forceinline__ float wave_sum32(float v) {
#pragma unroll
  for (int m = 16; m > 0; m >>= 1) v += __shfl_xor(v, m, 32);
  return v;
}

__global__ __launch_bounds__(NTHREADS)
void keypoint_encoder_kernel(const float* __restrict__ x,
                             const float* __restrict__ A,
                             const float* __restrict__ W1,
                             const float* __restrict__ g1,
                             const float* __restrict__ b1,
                             const float* __restrict__ W2,
                             const float* __restrict__ g2,
                             const float* __restrict__ b2,
                             const float* __restrict__ Wp,
                             const float* __restrict__ bp,
                             float* __restrict__ out)
{
  // region0: phases 0-3 hold sA (80x96 f16, 15360 B) + sH1T (128x96 f16, 24576 B)
  //          phases 4-6 reuse it as sZ2 (80x256 f16, 40960 B)
  __shared__ __align__(32) unsigned char smem0[40960];
  __shared__ __align__(32) _Float16 sAH[MP * MID];   // 20480 B
  __shared__ float sX[NJ * 3];
  __shared__ float sH0[NJ * 3];
  __shared__ float sMu[MP];
  __shared__ float sRstd[MP];
  __shared__ float sXm[3];

  _Float16* sA   = (_Float16*)smem0;             // [80][96]  rows i, cols j (K)
  _Float16* sH1T = (_Float16*)(smem0 + 15360);   // [128][96] rows o (N), cols j (K)
  _Float16* sZ2  = (_Float16*)smem0;             // [80][256]

  const int tid  = threadIdx.x;
  const int lane = tid & 31;
  const int wave = tid >> 5;
  const int hl   = (lane >= 16) ? 1 : 0;   // which half of the wave
  const int l16  = lane & 15;
  const int b    = blockIdx.x;

  // ---- Phase 0: zero padded region, stage x ----
  for (int i = tid; i < 40960 / 4; i += NTHREADS) ((unsigned int*)smem0)[i] = 0u;
  for (int t = tid; t < NJ * 3; t += NTHREADS) sX[t] = x[b * NJ * 3 + t];
  __syncthreads();

  // ---- Phase 1: A -> f16 LDS (padded); h0 = A@x; xmean for skip ----
  for (int idx = tid; idx < NJ * NJ; idx += NTHREADS) {
    int i = idx / NJ, j = idx % NJ;
    sA[i * KP + j] = (_Float16)A[idx];
  }
  if (tid < NJ * 3) {
    int i = tid / 3, c = tid % 3;
    float acc = 0.f;
    for (int j = 0; j < NJ; ++j) acc += A[i * NJ + j] * sX[j * 3 + c];
    sH0[tid] = acc;
  }
  if (tid < 3) {
    float s = 0.f;
    for (int j = 0; j < NJ; ++j) s += sX[j * 3 + tid];
    sXm[tid] = s * (1.f / NJ);
  }
  __syncthreads();

  // ---- Phase 2: z1 = h0 @ W1^T, LayerNorm(128) + ReLU -> sH1T (transposed, f16) ----
  for (int row = wave; row < NJ; row += 8) {
    float c0 = sH0[row * 3 + 0], c1 = sH0[row * 3 + 1], c2 = sH0[row * 3 + 2];
    float z[4]; float s = 0.f, ss = 0.f;
#pragma unroll
    for (int u = 0; u < 4; ++u) {
      int o = lane * 4 + u;
      float zv = W1[o * 3 + 0] * c0 + W1[o * 3 + 1] * c1 + W1[o * 3 + 2] * c2;
      z[u] = zv; s += zv; ss += zv * zv;
    }
    s = wave_sum32(s); ss = wave_sum32(ss);
    float mu = s * (1.f / MID);
    float var = ss * (1.f / MID) - mu * mu;
    float rstd = rsqrtf(var + 1e-5f);
#pragma unroll
    for (int u = 0; u < 4; ++u) {
      int o = lane * 4 + u;
      float v = (z[u] - mu) * rstd * g1[o] + b1[o];
      sH1T[o * KP + row] = (_Float16)fmaxf(v, 0.f);
    }
  }
  __syncthreads();

  // ---- Phase 3: AH = A @ H1 via WMMA f16 (M=80, K=96, N=128); one N-tile per wave ----
  {
    const int nt = wave;
    v16h bfrag[3];
#pragma unroll
    for (int ks = 0; ks < 3; ++ks) {
      // B-frag: lane holds column n; elements e = B[ks*32 + hl*16 + e][n] = sH1T[n][...]
      const _Float16* p = &sH1T[(nt * 16 + l16) * KP + ks * 32 + hl * 16];
      bfrag[ks] = *(const v16h*)p;
    }
    for (int mt = 0; mt < 5; ++mt) {
      v8f acc = {};
#pragma unroll
      for (int ks = 0; ks < 3; ++ks) {
        // A-frag: row = M, lane<16 K={0..15}, lane>=16 K={16..31}
        HV a;
        const int row = mt * 16 + l16;
        const int kLo = ks * 32 + hl * 16;
        a.v8[0] = *(const v8h*)&sA[row * KP + kLo];
        a.v8[1] = *(const v8h*)&sA[row * KP + kLo + 8];
        acc = __builtin_amdgcn_wmma_f32_16x16x32_f16(false, a.v16, false, bfrag[ks],
                                                     (short)0, acc, false, false);
      }
#pragma unroll
      for (int v = 0; v < 8; ++v) {
        int row = mt * 16 + v + hl * 8;
        sAH[row * MID + nt * 16 + l16] = (_Float16)acc[v];
      }
    }
  }
  __syncthreads();

  // ---- Phase 4: Z2 = AH @ W2^T via WMMA (M=80, K=128, N=256); 2 N-tiles per wave ----
  for (int p = 0; p < 2; ++p) {
    const int nt = wave + p * 8;
    const int n  = nt * 16 + l16;
    v16h bfrag[4];
#pragma unroll
    for (int ks = 0; ks < 4; ++ks) {
      // column n of W2^T is row n of W2: contiguous f32 -> f16
      const float* w2p = &W2[n * MID + ks * 32 + hl * 16];
      v16h bf;
#pragma unroll
      for (int e = 0; e < 16; ++e) bf[e] = (_Float16)w2p[e];
      bfrag[ks] = bf;
    }
    for (int mt = 0; mt < 5; ++mt) {
      v8f acc = {};
#pragma unroll
      for (int ks = 0; ks < 4; ++ks) {
        HV a;
        const int row = mt * 16 + l16;
        const int kLo = ks * 32 + hl * 16;
        a.v8[0] = *(const v8h*)&sAH[row * MID + kLo];
        a.v8[1] = *(const v8h*)&sAH[row * MID + kLo + 8];
        acc = __builtin_amdgcn_wmma_f32_16x16x32_f16(false, a.v16, false, bfrag[ks],
                                                     (short)0, acc, false, false);
      }
#pragma unroll
      for (int v = 0; v < 8; ++v) {
        int row = mt * 16 + v + hl * 8;
        sZ2[row * HID + nt * 16 + l16] = (_Float16)acc[v];
      }
    }
  }
  __syncthreads();

  // ---- Phase 5: LayerNorm(256) stats per joint row ----
  for (int row = wave; row < NJ; row += 8) {
    float s = 0.f, ss = 0.f;
#pragma unroll
    for (int u = 0; u < 8; ++u) {
      int o = lane * 8 + u;
      float zv = (float)sZ2[row * HID + o];
      s += zv; ss += zv * zv;
    }
    s = wave_sum32(s); ss = wave_sum32(ss);
    if (lane == 0) {
      float mu = s * (1.f / HID);
      float var = ss * (1.f / HID) - mu * mu;
      sMu[row] = mu;
      sRstd[row] = rsqrtf(var + 1e-5f);
    }
  }
  __syncthreads();

  // ---- Phase 6: out[o] = mean_i relu(LN(Z2)) + (Wp @ xmean + bp) ----
  {
    const int o = tid;
    const float g = g2[o], bb = b2[o];
    float acc = 0.f;
    for (int i = 0; i < NJ; ++i) {
      float zv = (float)sZ2[i * HID + o];
      float v = (zv - sMu[i]) * sRstd[i] * g + bb;
      acc += fmaxf(v, 0.f);
    }
    float skip = Wp[o * 3 + 0] * sXm[0] + Wp[o * 3 + 1] * sXm[1]
               + Wp[o * 3 + 2] * sXm[2] + bp[o];
    out[b * HID + o] = acc * (1.f / NJ) + skip;
  }
}

extern "C" void kernel_launch(void* const* d_in, const int* in_sizes, int n_in,
                              void* d_out, int out_size, void* d_ws, size_t ws_size,
                              hipStream_t stream) {
  const float* x  = (const float*)d_in[0];
  const float* A  = (const float*)d_in[1];
  const float* W1 = (const float*)d_in[2];
  const float* g1 = (const float*)d_in[3];
  const float* b1 = (const float*)d_in[4];
  const float* W2 = (const float*)d_in[5];
  const float* g2 = (const float*)d_in[6];
  const float* b2 = (const float*)d_in[7];
  const float* Wp = (const float*)d_in[8];
  const float* bp = (const float*)d_in[9];
  float* out = (float*)d_out;
  const int B = in_sizes[0] / (NJ * 3);
  keypoint_encoder_kernel<<<B, NTHREADS, 0, stream>>>(x, A, W1, g1, b1,
                                                      W2, g2, b2, Wp, bp, out);
}
